// ModulatedFourierFeatures_1520418423457
// MI455X (gfx1250) — matrix operand order repaired and checked by reading, not compiled, not run on hardware
//
#include <hip/hip_runtime.h>
#include <cstdint>

typedef __attribute__((ext_vector_type(16))) __bf16 v16bf;
typedef __attribute__((ext_vector_type(8)))  float  v8f;
typedef __attribute__((ext_vector_type(4)))  unsigned int v4u;
typedef __attribute__((ext_vector_type(8)))  int v8i;
typedef __attribute__((ext_vector_type(4)))  int v4i;

#define NFEAT   608        // [pos:32 | ctx:320 | x0:256] bf16 features per token
#define KA_T    11         // K-tiles for GEMM1 (352/32)
#define KB_T    18         // K-tiles for GEMM2 (576/32)
#define WAVES   4
#define WGT     128        // threads per workgroup (4 waves)
#define WG_TOK  64         // tokens per workgroup

// dynamic LDS carve (bytes)
#define SF_OFF   0
#define SF_BYTES (WAVES * 16 * NFEAT * 2)          // 77824
#define SW_OFF   (SF_OFF + SF_BYTES)               // 77824 (two 16 KB buffers)
#define SB1_OFF  (SW_OFF + 32768)                  // 110592
#define SB2_OFF  (SB1_OFF + 1024)                  // 111616
#define SWO_OFF  (SB2_OFF + 1024)                  // 112640
#define SMEM_BYTES (SWO_OFF + 1024)                // 113664

#define HAVE_TDM __has_builtin(__builtin_amdgcn_tensor_load_to_lds)

__device__ __forceinline__ unsigned short f2bf(float f) {
  unsigned u = __float_as_uint(f);
  unsigned r = u + 0x7FFFu + ((u >> 16) & 1u);   // round-to-nearest-even
  return (unsigned short)(r >> 16);
}

__device__ __forceinline__ unsigned lds_off(const void* p) {
  return (unsigned)(uintptr_t)p;                 // generic LDS addr low 32b = offset
}

#if HAVE_TDM
// Issue one TDM copy: 16 KB contiguous global -> LDS. D# per ISA 8.3/8.4:
// 1-D tile, data_size=8B (code 3), tile_dim0 = tensor_dim0 = stride = 2048 units.
__device__ __forceinline__ void tdm_stage(const unsigned short* __restrict__ src,
                                          unsigned lds_byte_off) {
  const unsigned long long ga = (unsigned long long)(uintptr_t)src;
  const v4u g0 = { 1u,                                            // count=1, user desc
                   lds_byte_off,                                  // lds_addr
                   (unsigned)(ga & 0xFFFFFFFFull),                // global_addr[31:0]
                   ((unsigned)(ga >> 32) & 0x01FFFFFFu) | 0x80000000u }; // [56:32] | type=2
  const v8i g1 = { (int)(3u << 16),        // data_size=3 (8B), wg_mask=0, no pad/iterate
                   (int)(2048u << 16),     // tensor_dim0[15:0] @ bits 63:48
                   (int)(1u << 16),        // tensor_dim0[31:16]=0, tensor_dim1[15:0]=1
                   (int)(2048u << 16),     // tensor_dim1[31:16]=0, tile_dim0=2048
                   0,                      // tile_dim1=0, tile_dim2=0 (1-D)
                   2048,                   // tensor_dim0_stride[31:0]
                   0, 0 };                 // stride hi / tensor_dim1_stride
  const v4i z4 = { 0, 0, 0, 0 };
  const v8i z8 = { 0, 0, 0, 0, 0, 0, 0, 0 };
  __builtin_amdgcn_tensor_load_to_lds(g0, g1, z4, z4, z8, 0);
}
#endif

// ---------------------------------------------------------------------------
// Prep 1: pre-swizzle weights into bf16 WMMA A-fragments + fused biases.
//   lane<16 : elem e -> K = e                      M = lane
//   lane>=16: elem e -> K = e+16                   M = lane-16
// Fragment storage: frag*512 + lane*16 + e (ushort).
// ---------------------------------------------------------------------------
__global__ void prep_weights(const float* __restrict__ W_mod, const float* __restrict__ W0,
                             const float* __restrict__ W1,
                             const float* __restrict__ b_mod, const float* __restrict__ b0,
                             const float* __restrict__ b1,
                             unsigned short* __restrict__ waf, unsigned short* __restrict__ wbf,
                             float* __restrict__ bias1, float* __restrict__ bias2) {
  const int id = blockIdx.x * 256 + threadIdx.x;
  if (id < 256) {
    bias1[id] = b0[id] + b_mod[id];
    bias2[id] = b1[id] + b_mod[256 + id];
  }
  const int e = id & 15, l = (id >> 4) & 31, frag = id >> 9;
  const int m = frag & 15, kt = frag >> 4;
  const int M = m * 16 + (l & 15);
  const int K = kt * 32 + e + ((l >= 16) ? 16 : 0);

  if (id < KA_T * 16 * 512) {            // GEMM1: Wa = [W0 | W_mod[:256]] (256 x 352)
    float v = 0.0f;
    if (K < 32) { if (K < 17) v = W0[M * 17 + K]; }
    else { const int c = K - 32; if (c < 296) v = W_mod[M * 296 + c]; }
    waf[id] = f2bf(v);
  }
  if (id < KB_T * 16 * 512) {            // GEMM2: Wb = [W1 | W_mod[256:]] (256 x 576)
    float v = 0.0f;
    if (K < 256) v = W1[M * 256 + K];
    else { const int c = K - 256; if (c < 296) v = W_mod[(256 + M) * 296 + c]; }
    wbf[id] = f2bf(v);
  }
}

// ---------------------------------------------------------------------------
// Prep 2: per-batch static encoder -> static_emb (B x 40) fp32.
// ---------------------------------------------------------------------------
__global__ void prep_statics(const float* __restrict__ xs, const int* __restrict__ dir_idx,
                             const float* __restrict__ Bmat,
                             const float* __restrict__ Ws1, const float* __restrict__ bs1,
                             const float* __restrict__ Ws2, const float* __restrict__ bs2,
                             const float* __restrict__ dir_emb, float* __restrict__ semb) {
  const int b = blockIdx.x * 64 + threadIdx.x;
  if (b >= 1024) return;
  const float x0 = xs[b * 2 + 0], x1 = xs[b * 2 + 1];
  float feats[32];
  #pragma unroll
  for (int j = 0; j < 16; ++j) {
    const float p = 6.283185307179586f * (x0 * Bmat[j] + x1 * Bmat[16 + j]);
    feats[j] = __sinf(p);
    feats[16 + j] = __cosf(p);
  }
  float h[64];
  for (int o = 0; o < 64; ++o) {
    float s = bs1[o];
    #pragma unroll 8
    for (int i = 0; i < 32; ++i) s += feats[i] * Ws1[o * 32 + i];
    h[o] = 0.5f * s * (1.0f + erff(s * 0.70710678118654752f));   // exact GELU
  }
  for (int o = 0; o < 32; ++o) {
    float s = bs2[o];
    #pragma unroll 8
    for (int i = 0; i < 64; ++i) s += h[i] * Ws2[o * 64 + i];
    semb[b * 40 + o] = s;
  }
  const int d = dir_idx[b];
  #pragma unroll
  for (int j = 0; j < 8; ++j) semb[b * 40 + 32 + j] = dir_emb[d * 8 + j];
}

// ---------------------------------------------------------------------------
// One GEMM pass: KT K-steps, 16 m-tiles, TDM double-buffered weight staging.
// Buffer selection is done with an index offset off a SINGLE LDS base pointer
// so address-space inference keeps A-fragment reads on the ds_load path.
// ---------------------------------------------------------------------------
template <int KT, bool G2, bool PRE>
__device__ __forceinline__ void gemm_pass(v8f (&acc)[16],
                                          const unsigned short* __restrict__ wfrag,
                                          unsigned short* sW,     // 2 x 8192 ushorts
                                          const unsigned short* sBrow,
                                          int tid, int wave, int lane) {
#if HAVE_TDM
  const unsigned sw_base = lds_off(sW);
  if (!PRE) {
    __syncthreads();                               // prior consumers of bufs done
    if (wave == 0) tdm_stage(wfrag, sw_base);
  }
  for (int kt = 0; kt < KT; ++kt) {
    __builtin_amdgcn_s_wait_tensorcnt(0);          // buffer kt landed (wave0; others no-op)
    __syncthreads();
    if (wave == 0 && kt + 1 < KT)
      tdm_stage(wfrag + (size_t)(kt + 1) * 8192, sw_base + (((kt + 1) & 1) << 14));
    const unsigned buf = (kt & 1) * 8192;          // ushort offset, same base pointer
    const int fb = G2 ? ((kt < 8) ? (352 + kt * 32) : (32 + (kt - 8) * 32)) : (kt * 32);
    const v16bf bfrag = *(const v16bf*)(sBrow + fb);
    #pragma unroll
    for (int m = 0; m < 16; ++m) {
      const v16bf afrag = *(const v16bf*)(sW + buf + (m * 32 + lane) * 16);
      acc[m] = __builtin_amdgcn_wmma_f32_16x16x32_bf16(false, afrag, false, bfrag,
                                                       (short)0, acc[m], false, false);
    }
  }
#else
  // synchronous fallback staging (no TDM builtin on this toolchain)
  for (int kt = 0; kt < KT; ++kt) {
    __syncthreads();
    const uint4* src = (const uint4*)(wfrag + (size_t)kt * 8192);
    uint4* dst = (uint4*)sW;
    #pragma unroll
    for (int i = 0; i < 8; ++i) dst[i * WGT + tid] = src[i * WGT + tid];
    __syncthreads();
    const int fb = G2 ? ((kt < 8) ? (352 + kt * 32) : (32 + (kt - 8) * 32)) : (kt * 32);
    const v16bf bfrag = *(const v16bf*)(sBrow + fb);
    #pragma unroll
    for (int m = 0; m < 16; ++m) {
      const v16bf afrag = *(const v16bf*)(sW + (m * 32 + lane) * 16);
      acc[m] = __builtin_amdgcn_wmma_f32_16x16x32_bf16(false, afrag, false, bfrag,
                                                       (short)0, acc[m], false, false);
    }
  }
#endif
}

// ---------------------------------------------------------------------------
// Main fused kernel: NeRF encode + ctx assembly + two WMMA GEMMs + output dot.
// ---------------------------------------------------------------------------
__global__ __launch_bounds__(WGT)
void film_main(const float* __restrict__ coords, const float* __restrict__ code,
               const float* __restrict__ hs, const float* __restrict__ semb,
               const unsigned short* __restrict__ waf, const unsigned short* __restrict__ wbf,
               const float* __restrict__ bias1, const float* __restrict__ bias2,
               const float* __restrict__ Wo, const float* __restrict__ bo,
               float* __restrict__ out) {
  extern __shared__ __align__(128) char smem[];
  unsigned short* sF = (unsigned short*)(smem + SF_OFF);
  unsigned short* sW = (unsigned short*)(smem + SW_OFF);
  float* sB1 = (float*)(smem + SB1_OFF);
  float* sB2 = (float*)(smem + SB2_OFF);
  float* sWo = (float*)(smem + SWO_OFF);

  const int tid  = threadIdx.x;
  const int lane = tid & 31;
  const int wave = tid >> 5;
  const int half = lane >> 4;
  const int tok  = lane & 15;
  const int gtok = blockIdx.x * WG_TOK + wave * 16 + tok;
  const int b    = gtok >> 9;      // T = 512

#if HAVE_TDM
  if (wave == 0) tdm_stage(waf, lds_off(sW));    // prefetch GEMM1 kt=0 under fill phase
#endif

  for (int i = tid; i < 256; i += WGT) { sB1[i] = bias1[i]; sB2[i] = bias2[i]; sWo[i] = Wo[i]; }

  // ---- assemble per-token feature row: [pos(32) | ctx(320)] ----
  unsigned short* Ft = &sF[(wave * 16 + tok) * NFEAT];
  if (half == 0) {
    const float c0 = coords[gtok];
    Ft[0] = f2bf(c0);
    #pragma unroll
    for (int j = 0; j < 8; ++j) {
      const float w = c0 * (1.25f * (float)j);
      Ft[1 + j] = f2bf(__sinf(w));
      Ft[9 + j] = f2bf(__cosf(w));
    }
    #pragma unroll
    for (int j = 17; j < 32; ++j) Ft[j] = 0;
    const float4* cd = (const float4*)(code + (size_t)b * 128);
    #pragma unroll 4
    for (int j = 0; j < 32; ++j) {
      const float4 v = cd[j]; const int o = 32 + 4 * j;
      Ft[o] = f2bf(v.x); Ft[o + 1] = f2bf(v.y); Ft[o + 2] = f2bf(v.z); Ft[o + 3] = f2bf(v.w);
    }
  } else {
    const float4* hp = (const float4*)(hs + (size_t)gtok * 128);
    #pragma unroll 4
    for (int j = 0; j < 32; ++j) {
      const float4 v = hp[j]; const int o = 160 + 4 * j;
      Ft[o] = f2bf(v.x); Ft[o + 1] = f2bf(v.y); Ft[o + 2] = f2bf(v.z); Ft[o + 3] = f2bf(v.w);
    }
    const float4* sp = (const float4*)(semb + (size_t)b * 40);
    #pragma unroll
    for (int j = 0; j < 10; ++j) {
      const float4 v = sp[j]; const int o = 288 + 4 * j;
      Ft[o] = f2bf(v.x); Ft[o + 1] = f2bf(v.y); Ft[o + 2] = f2bf(v.z); Ft[o + 3] = f2bf(v.w);
    }
    #pragma unroll
    for (int j = 328; j < 352; ++j) Ft[j] = 0;
  }

  const unsigned short* sBrow = &sF[(wave * 16 + tok) * NFEAT + half * 16];

  v8f acc[16];
  #pragma unroll
  for (int m = 0; m < 16; ++m) acc[m] = (v8f){0.f, 0.f, 0.f, 0.f, 0.f, 0.f, 0.f, 0.f};

  // ---- GEMM1: x0 = relu([W0|Wmod_lo] @ [pos;ctx] + bias1) ----
  gemm_pass<KA_T, false, true>(acc, waf, sW, sBrow, tid, wave, lane);
  {
    const int hi8 = half * 8;
    #pragma unroll
    for (int m = 0; m < 16; ++m) {
      #pragma unroll
      for (int r = 0; r < 8; ++r) {
        float v = acc[m][r] + sB1[m * 16 + hi8 + r];
        v = fmaxf(v, 0.0f);
        sF[(wave * 16 + tok) * NFEAT + 352 + m * 16 + hi8 + r] = f2bf(v);
      }
    }
  }

  // ---- GEMM2: x1 = relu([W1|Wmod_hi] @ [x0;ctx] + bias2) ----
  #pragma unroll
  for (int m = 0; m < 16; ++m) acc[m] = (v8f){0.f, 0.f, 0.f, 0.f, 0.f, 0.f, 0.f, 0.f};
  gemm_pass<KB_T, true, false>(acc, wbf, sW, sBrow, tid, wave, lane);

  // ---- output: out = Wo . x1 + bo, reduced across the two lane halves ----
  float oacc = 0.0f;
  const int hi8o = half * 8;
  #pragma unroll
  for (int m = 0; m < 16; ++m) {
    #pragma unroll
    for (int r = 0; r < 8; ++r) {
      const float v = fmaxf(acc[m][r] + sB2[m * 16 + hi8o + r], 0.0f);
      oacc += v * sWo[m * 16 + hi8o + r];
    }
  }
  oacc += __shfl_xor(oacc, 16, 32);
  if (half == 0) out[gtok] = oacc + bo[0];
}

// ---------------------------------------------------------------------------
extern "C" void kernel_launch(void* const* d_in, const int* in_sizes, int n_in,
                              void* d_out, int out_size, void* d_ws, size_t ws_size,
                              hipStream_t stream) {
  const float* coords  = (const float*)d_in[0];
  const float* code    = (const float*)d_in[1];
  const float* hs      = (const float*)d_in[2];
  const float* xs      = (const float*)d_in[3];
  const int*   dir_idx = (const int*)d_in[4];
  const float* W_mod   = (const float*)d_in[5];
  const float* b_mod   = (const float*)d_in[6];
  const float* W0      = (const float*)d_in[7];
  const float* b0      = (const float*)d_in[8];
  const float* W1      = (const float*)d_in[9];
  const float* b1      = (const float*)d_in[10];
  const float* Wo      = (const float*)d_in[11];
  const float* bo      = (const float*)d_in[12];
  const float* Bmat    = (const float*)d_in[13];
  const float* Ws1     = (const float*)d_in[14];
  const float* bs1     = (const float*)d_in[15];
  const float* Ws2     = (const float*)d_in[16];
  const float* bs2     = (const float*)d_in[17];
  const float* dir_emb = (const float*)d_in[18];

  char* ws = (char*)d_ws;
  unsigned short* waf   = (unsigned short*)(ws);                 // 180224 B
  unsigned short* wbf   = (unsigned short*)(ws + 180224);        // 294912 B
  float*          bias1 = (float*)(ws + 475136);
  float*          bias2 = (float*)(ws + 476160);
  float*          semb  = (float*)(ws + 477184);                 // 163840 B

  (void)hipFuncSetAttribute((const void*)film_main,
                            hipFuncAttributeMaxDynamicSharedMemorySize, SMEM_BYTES);

  prep_weights<<<576, 256, 0, stream>>>(W_mod, W0, W1, b_mod, b0, b1, waf, wbf, bias1, bias2);
  prep_statics<<<16, 64, 0, stream>>>(xs, dir_idx, Bmat, Ws1, bs1, Ws2, bs2, dir_emb, semb);

  const int n_tok = 1024 * 512;
  film_main<<<n_tok / WG_TOK, WGT, SMEM_BYTES, stream>>>(coords, code, hs, semb, waf, wbf,
                                                         bias1, bias2, Wo, bo, (float*)d_out);
}
